// SlidingWindowAttention_86380382257703
// MI455X (gfx1250) — matrix operand order repaired and checked by reading, not compiled, not run on hardware
//
#include <hip/hip_runtime.h>

typedef __attribute__((ext_vector_type(16))) _Float16 v16h;
typedef __attribute__((ext_vector_type(8)))  _Float16 v8h;
typedef __attribute__((ext_vector_type(4)))  _Float16 v4h;
typedef __attribute__((ext_vector_type(8)))  float    v8f;

#define B_   2
#define S_   2048
#define D_   2048
#define H_   16
#define DH_  128
#define WIN_ 512

// DPP row_xmask reduction step: xor-exchange within each 16-lane row (pure VALU, no LDS pipe)
template<int CTRL>
__device__ __forceinline__ float dpp_xor_max(float v) {
  int t = __builtin_amdgcn_update_dpp(0, __builtin_bit_cast(int, v), CTRL, 0xf, 0xf, true);
  return fmaxf(v, __builtin_bit_cast(float, t));
}
__device__ __forceinline__ float rowmax16(float v) {
  v = dpp_xor_max<0x161>(v);   // row_xmask:1
  v = dpp_xor_max<0x162>(v);   // row_xmask:2
  v = dpp_xor_max<0x164>(v);   // row_xmask:4
  v = dpp_xor_max<0x168>(v);   // row_xmask:8
  return v;
}

// ---------------------------------------------------------------- fp32 -> fp16
__global__ __launch_bounds__(256) void cvt_f32_f16(const float* __restrict__ src,
                                                   _Float16* __restrict__ dst, int n) {
  int i = (blockIdx.x * blockDim.x + threadIdx.x) * 4;
  if (i + 3 < n) {
    float4 v = *(const float4*)(src + i);
    v4h o; o[0] = (_Float16)v.x; o[1] = (_Float16)v.y; o[2] = (_Float16)v.z; o[3] = (_Float16)v.w;
    *(v4h*)(dst + i) = o;
  }
}

// ---------------------------------------------------------------- WMMA GEMM
// C[m,n] = sum_k A[m,k] * Bw[n,k]    (A: MxK row-major f16, Bw: NxK row-major f16)
// One wave computes a 64x32 tile of C (8 WMMAs per 32-wide K step).
template<bool F32OUT>
__global__ __launch_bounds__(256) void gemm_wmma(const _Float16* __restrict__ A,
                                                 const _Float16* __restrict__ Bw,
                                                 _Float16* __restrict__ Cf16,
                                                 float* __restrict__ Cf32,
                                                 int M, int N, int K) {
  const int lane = threadIdx.x & 31;
  const int wave = threadIdx.x >> 5;
  const int tilesN = N >> 5;
  const int w    = blockIdx.x * 8 + wave;
  const int row0 = (w / tilesN) << 6;   // 64 rows
  const int col0 = (w % tilesN) << 5;   // 32 cols
  const int r16 = lane & 15, hf = lane >> 4;

  v8f acc[4][2] = {};
  for (int kb = 0; kb < K; kb += 32) {
    v16h a[4], b[2];
#pragma unroll
    for (int mi = 0; mi < 4; ++mi) {
      // A fragment, contiguous k order used here: lanes 0-15 hold K 0..15, lanes 16-31 K 16..31
      const _Float16* pa = A + (size_t)(row0 + mi * 16 + r16) * K + kb;
      a[mi] = *(const v16h*)(pa + hf * 16);
      __builtin_prefetch(pa + 64, 0, 1);
    }
#pragma unroll
    for (int ni = 0; ni < 2; ++ni) {
      // B fragment, same contiguous k order as A: lanes 0-15 col N=lane K=0..15; lanes 16-31 col N=lane-16 K=16..31
      const _Float16* pb = Bw + (size_t)(col0 + ni * 16 + r16) * K + kb + hf * 16;
      b[ni] = *(const v16h*)pb;
    }
#pragma unroll
    for (int mi = 0; mi < 4; ++mi)
#pragma unroll
      for (int ni = 0; ni < 2; ++ni)
        acc[mi][ni] = __builtin_amdgcn_wmma_f32_16x16x32_f16(
            false, a[mi], false, b[ni], (short)0, acc[mi][ni], false, false);
  }
#pragma unroll
  for (int mi = 0; mi < 4; ++mi)
#pragma unroll
    for (int ni = 0; ni < 2; ++ni)
#pragma unroll
      for (int r = 0; r < 8; ++r) {
        int gr = row0 + mi * 16 + r + hf * 8;   // C layout: M = r + 8*(lane>=16)
        int gc = col0 + ni * 16 + r16;          //           N = lane & 15
        if (F32OUT) Cf32[(size_t)gr * N + gc] = acc[mi][ni][r];
        else        Cf16[(size_t)gr * N + gc] = (_Float16)acc[mi][ni][r];
      }
}

// ---------------------------------------------------------------- RoPE + head scatter
// q,k: [B*S, D] -> [B,H,S,Dh] with RoPE.  v: [B*S, D] -> transposed [B,H,Dh,S].
__global__ __launch_bounds__(256) void rope_scatter(const _Float16* __restrict__ qt,
                                                    const _Float16* __restrict__ kt,
                                                    const _Float16* __restrict__ vt0,
                                                    _Float16* __restrict__ qr,
                                                    _Float16* __restrict__ kr,
                                                    _Float16* __restrict__ vtt) {
  int idx = blockIdx.x * blockDim.x + threadIdx.x;   // ((b*S+s)*H+h)*64+u
  int u = idx & 63;
  int h = (idx >> 6) & (H_ - 1);
  int s = (idx >> 10) & (S_ - 1);
  int b = idx >> 21;
  size_t in = (size_t)(b * S_ + s) * D_ + h * DH_ + 2 * u;
  float q1 = (float)qt[in],  q2 = (float)qt[in + 1];
  float k1 = (float)kt[in],  k2 = (float)kt[in + 1];
  float v1 = (float)vt0[in], v2 = (float)vt0[in + 1];
  float inv = __powf(10000.0f, -(float)u * (1.0f / 64.0f));
  float ang = (float)s * inv;
  float sn, cs; __sincosf(ang, &sn, &cs);
  size_t oq = (size_t)((b * H_ + h) * S_ + s) * DH_ + 2 * u;
  qr[oq]     = (_Float16)(q1 * cs - q2 * sn);
  qr[oq + 1] = (_Float16)(q1 * sn + q2 * cs);
  kr[oq]     = (_Float16)(k1 * cs - k2 * sn);
  kr[oq + 1] = (_Float16)(k1 * sn + k2 * cs);
  size_t ov = (size_t)((b * H_ + h) * DH_ + 2 * u) * S_ + s;
  vtt[ov]      = (_Float16)v1;
  vtt[ov + S_] = (_Float16)v2;
}

// ---------------------------------------------------------------- one 16x16 key tile
// MASKED only for the two edge tiles (diagonal and, when qbase>=512, the oldest tile).
template<bool MASKED>
__device__ __forceinline__ void attn_tile(int kb, int qbase, int r16, int hf,
                                          const v16h* aq, v16h bone,
                                          const _Float16* __restrict__ khead,
                                          const _Float16* __restrict__ vhead,
                                          _Float16* pb, v8f* O,
                                          float* mrow, float* lrow) {
  // Issue all K and V fragment loads up front (distinct regs -> one clause, staggered waits;
  // V loads overlap the whole softmax chain).
  v16h bk[4];
#pragma unroll
  for (int kc = 0; kc < 4; ++kc)
    bk[kc] = *(const v16h*)(khead + (size_t)(kb + r16) * DH_ + kc * 32 + hf * 16);
  v16h bv[8];
#pragma unroll
  for (int dt = 0; dt < 8; ++dt)
    bv[dt] = *(const v16h*)(vhead + (size_t)(dt * 16 + r16) * S_ + kb + hf * 16);

  // ---- scores S = Q * K^T  (16x16, K-dim = 128 via 4 WMMAs)
  v8f s = {};
#pragma unroll
  for (int kc = 0; kc < 4; ++kc)
    s = __builtin_amdgcn_wmma_f32_16x16x32_f16(false, aq[kc], false, bk[kc], (short)0, s, false, false);

  const float scale = 0.08838834764831845f;  // 128^-0.5
  int kpos = kb + r16;
  float alpha[8];
#pragma unroll
  for (int r = 0; r < 8; ++r) {
    float sv;
    bool keep = true;
    if (MASKED) {
      int qpos = qbase + r + hf * 8;
      keep = (unsigned)(qpos - kpos) < (unsigned)WIN_;
      sv = keep ? s[r] * scale : -1e30f;
    } else {
      sv = s[r] * scale;
    }
    float mnew = fmaxf(mrow[r], rowmax16(sv));
    alpha[r] = __expf(mrow[r] - mnew);
    mrow[r] = mnew;
    float pv = MASKED ? (keep ? __expf(sv - mnew) : 0.0f) : __expf(sv - mnew);
    pb[(r + hf * 8) * 16 + r16] = (_Float16)pv;   // transpose P via LDS
  }
  // per-wave LDS ordering: wait for our own ds stores before cross-lane read
  asm volatile("s_wait_dscnt 0x0" ::: "memory");
  v16h ap;
  {
    v16h lo = *(const v16h*)(pb + r16 * 16);   // A fragment, contiguous k order: lanes 0-15 K=0..15; lanes 16-31 K=16..31 = 0
    v16h z  = {};
    ap = hf ? z : lo;
  }
  // ---- row sums of P via matrix unit: each lane gets its 8 row-sums in the C fragment
  v8f lsum = {};
  lsum = __builtin_amdgcn_wmma_f32_16x16x32_f16(false, ap, false, bone, (short)0, lsum, false, false);
#pragma unroll
  for (int r = 0; r < 8; ++r) {
    lrow[r] = lrow[r] * alpha[r] + lsum[r];
#pragma unroll
    for (int t = 0; t < 8; ++t) O[t][r] *= alpha[r];
  }
  // ---- O += P * V   (V pre-transposed: column d contiguous over s)
#pragma unroll
  for (int dt = 0; dt < 8; ++dt)
    O[dt] = __builtin_amdgcn_wmma_f32_16x16x32_f16(false, ap, false, bv[dt], (short)0, O[dt], false, false);
}

// ---------------------------------------------------------------- flash attention (window 512)
// One wave per (b, h, 16-query tile). WMMA for QK^T, P*V, and P row-sums (P x ones).
__global__ __launch_bounds__(256) void attn_wmma(const _Float16* __restrict__ qr,
                                                 const _Float16* __restrict__ kr,
                                                 const _Float16* __restrict__ vt,
                                                 _Float16* __restrict__ ao) {
  __shared__ __align__(32) _Float16 pbuf[8][16][16];  // per-wave P tile
  const int lane = threadIdx.x & 31;
  const int wv   = threadIdx.x >> 5;
  const int w    = blockIdx.x * 8 + wv;
  const int qt   = w & (S_ / 16 - 1);
  const int h    = (w / (S_ / 16)) & (H_ - 1);
  const int b    = w / (S_ / 16 * H_);
  const int qbase = qt * 16;
  const int r16 = lane & 15, hf = lane >> 4;

  const _Float16* qhead = qr + (size_t)(b * H_ + h) * S_ * DH_;
  const _Float16* khead = kr + (size_t)(b * H_ + h) * S_ * DH_;
  const _Float16* vhead = vt + (size_t)(b * H_ + h) * DH_ * S_;
  _Float16* pb = &pbuf[wv][0][0];

  v16h aq[4];
  {
    const _Float16* pq = qhead + (size_t)(qbase + r16) * DH_;
#pragma unroll
    for (int kc = 0; kc < 4; ++kc) {
      aq[kc] = *(const v16h*)(pq + kc * 32 + hf * 16);
    }
  }

  v16h bone;   // all-ones B fragment: rowsum(P) = P x ones via one extra WMMA
#pragma unroll
  for (int e = 0; e < 16; ++e) bone[e] = (_Float16)1.0f;

  v8f O[8] = {};
  float mrow[8], lrow[8];
#pragma unroll
  for (int r = 0; r < 8; ++r) { mrow[r] = -1e30f; lrow[r] = 0.0f; }

  // Tile schedule: [oldest edge tile (masked)] [interior tiles (no mask)] [diagonal (masked)]
  int kb = qbase - WIN_ + 1;
  if (kb < 0) kb = 0;
  kb &= ~15;                       // == qbase-512 when qbase >= 512, else 0
  if (qbase >= WIN_) {
    attn_tile<true>(kb, qbase, r16, hf, aq, bone, khead, vhead, pb, O, mrow, lrow);
    kb += 16;
  }
  for (; kb < qbase; kb += 16)
    attn_tile<false>(kb, qbase, r16, hf, aq, bone, khead, vhead, pb, O, mrow, lrow);
  attn_tile<true>(qbase, qbase, r16, hf, aq, bone, khead, vhead, pb, O, mrow, lrow);

  // ---- normalize + scatter back to [B,S,D] token-major f16
  _Float16* aop = ao + (size_t)b * S_ * D_;
#pragma unroll
  for (int r = 0; r < 8; ++r) {
    float invl = 1.0f / lrow[r];
#pragma unroll
    for (int dt = 0; dt < 8; ++dt) {
      int row = qbase + r + hf * 8;
      int col = h * DH_ + dt * 16 + r16;
      aop[(size_t)row * D_ + col] = (_Float16)(O[dt][r] * invl);
    }
  }
}

// ---------------------------------------------------------------- host
extern "C" void kernel_launch(void* const* d_in, const int* in_sizes, int n_in,
                              void* d_out, int out_size, void* d_ws, size_t ws_size,
                              hipStream_t stream) {
  const float* x  = (const float*)d_in[0];
  const float* Wq = (const float*)d_in[1];
  const float* Wk = (const float*)d_in[2];
  const float* Wv = (const float*)d_in[3];
  const float* Wo = (const float*)d_in[4];
  float* out = (float*)d_out;

  const size_t SXD = (size_t)B_ * S_ * D_;   // 8,388,608
  const size_t WD  = (size_t)D_ * D_;        // 4,194,304
  _Float16* ws  = (_Float16*)d_ws;
  _Float16* xh  = ws;
  _Float16* wqh = xh  + SXD;
  _Float16* wkh = wqh + WD;
  _Float16* wvh = wkh + WD;
  _Float16* woh = wvh + WD;
  _Float16* qt  = woh + WD;
  _Float16* kt  = qt  + SXD;
  _Float16* vt0 = kt  + SXD;
  _Float16* qr  = vt0 + SXD;
  _Float16* kr  = qr  + SXD;
  _Float16* vtt = kr  + SXD;
  _Float16* ao  = vtt + SXD;

  const int M = B_ * S_;   // 4096
  cvt_f32_f16<<<(int)(SXD / 1024), 256, 0, stream>>>(x,  xh,  (int)SXD);
  cvt_f32_f16<<<(int)(WD  / 1024), 256, 0, stream>>>(Wq, wqh, (int)WD);
  cvt_f32_f16<<<(int)(WD  / 1024), 256, 0, stream>>>(Wk, wkh, (int)WD);
  cvt_f32_f16<<<(int)(WD  / 1024), 256, 0, stream>>>(Wv, wvh, (int)WD);
  cvt_f32_f16<<<(int)(WD  / 1024), 256, 0, stream>>>(Wo, woh, (int)WD);

  const int gemmBlocks = (M / 64) * (D_ / 32) / 8;   // 512
  gemm_wmma<false><<<gemmBlocks, 256, 0, stream>>>(xh, wqh, qt,  nullptr, M, D_, D_);
  gemm_wmma<false><<<gemmBlocks, 256, 0, stream>>>(xh, wkh, kt,  nullptr, M, D_, D_);
  gemm_wmma<false><<<gemmBlocks, 256, 0, stream>>>(xh, wvh, vt0, nullptr, M, D_, D_);

  rope_scatter<<<(B_ * S_ * H_ * 64) / 256, 256, 0, stream>>>(qt, kt, vt0, qr, kr, vtt);

  attn_wmma<<<(B_ * H_ * (S_ / 16)) / 8, 256, 0, stream>>>(qr, kr, vtt, ao);

  gemm_wmma<true><<<gemmBlocks, 256, 0, stream>>>(ao, woh, nullptr, out, M, D_, D_);
}
